// SDARAttention_64355789963877
// MI455X (gfx1250) — matrix operand order repaired and checked by reading, not compiled, not run on hardware
//
#include <hip/hip_runtime.h>

typedef __bf16 bf16_t;
typedef __attribute__((ext_vector_type(16))) __bf16 v16bf;
typedef __attribute__((ext_vector_type(8)))  __bf16 v8bf;
typedef __attribute__((ext_vector_type(8)))  float  v8f;

constexpr int   T_      = 3072;
constexpr int   HIDDEN_ = 2048;
constexpr int   NH_     = 16;
constexpr int   NKV_    = 4;
constexpr int   HD_     = 128;
constexpr int   QKVO_   = (NH_ + 2 * NKV_) * HD_;   // 3072
constexpr int   WINDOW_ = 1024;
constexpr float SCALE_  = 0.08838834764831843f;     // 128^-0.5

constexpr int   LDSPITCH_ = 40;                      // 32 + 8 pad (80B = 20 banks)

__device__ __forceinline__ v8f wmma_bf16(v16bf a, v16bf b, v8f c) {
    return __builtin_amdgcn_wmma_f32_16x16x32_bf16(
        false, a, false, b, (short)0, c, false, false);
}

// Per-lane async 16B copy global -> LDS (tracked by ASYNCcnt).
// LDS operand: wave-relative offset = low 32 bits of the generic LDS address.
__device__ __forceinline__ void async_b128(void* lds_dst, const void* g_src) {
    unsigned int l = (unsigned int)(unsigned long long)lds_dst;
    asm volatile("global_load_async_to_lds_b128 %0, %1, off"
                 :: "v"(l), "v"(g_src) : "memory");
}
__device__ __forceinline__ void wait_async0() {
    asm volatile("s_wait_asynccnt 0x0" ::: "memory");
}

// ---------------------------------------------------------------- conversion
__global__ void f32_to_bf16_kernel(const float* __restrict__ in,
                                   bf16_t* __restrict__ out, int n) {
    int i = blockIdx.x * blockDim.x + threadIdx.x;
    int stride = gridDim.x * blockDim.x;
    for (; i < n; i += stride) out[i] = (bf16_t)in[i];
}

// ---------------------------------------------------------------- GEMM (bf16 WMMA, f32 out)
// C[M,N] = A[M,K] @ B[N,K]^T  (row-major bf16), C fp32 row-major.
// Block = 256 threads (8 waves) computes a 128x128 tile.
// A (128x32) and B (128x32) staged in LDS via async global->LDS copies,
// double-buffered; wave (wm,wn) computes 32 rows x 64 cols = 8 WMMAs/K-step.
__global__ __launch_bounds__(256)
void gemm_bf16_kernel(const bf16_t* __restrict__ A, const bf16_t* __restrict__ B,
                      float* __restrict__ C, int M, int N, int K) {
    __shared__ __align__(16) bf16_t smA[2][128 * LDSPITCH_];
    __shared__ __align__(16) bf16_t smB[2][128 * LDSPITCH_];

    const int tid   = threadIdx.x;
    const int lane  = tid & 31;
    const int wave  = tid >> 5;
    const int wm    = wave >> 1;          // 0..3 -> 32-row strip
    const int wn    = wave & 1;           // 0..1 -> 64-col strip
    const int m     = lane & 15;
    const int half  = lane >> 4;
    const int kbA   = half ? 8 : 0;
    const int kbB   = half ? 16 : 0;

    const int nBase = blockIdx.x * 128;
    const int mBase = blockIdx.y * 128;

    // Stage one 128x32 K-slice of A and B into LDS buffer b (async).
    auto stage = [&](int ks, int b) {
#pragma unroll
        for (int i = 0; i < 2; ++i) {
            const int c = tid + 256 * i;            // 512 chunks of 16B per tile
            const int r = c >> 2;                   // row 0..127
            const int p = c & 3;                    // 16B part within 64B row-slice
            const int k0 = ks * 32 + p * 8;
            async_b128((void*)(smA[b] + r * LDSPITCH_ + p * 8),
                       A + (size_t)(mBase + r) * K + k0);
            async_b128((void*)(smB[b] + r * LDSPITCH_ + p * 8),
                       B + (size_t)(nBase + r) * K + k0);
        }
    };

    v8f acc[2][4] = {};
    const int nk = K >> 5;

    stage(0, 0);
    wait_async0();
    __syncthreads();

    for (int ks = 0; ks < nk; ++ks) {
        if (ks + 1 < nk) stage(ks + 1, (ks + 1) & 1);   // overlap with compute

        const bf16_t* bufA = smA[ks & 1];
        const bf16_t* bufB = smB[ks & 1];

        union { v16bf v; v8bf h[2]; } af[2];
#pragma unroll
        for (int mi = 0; mi < 2; ++mi) {
            const bf16_t* ar = bufA + (wm * 32 + mi * 16 + m) * LDSPITCH_;
            af[mi].h[0] = *(const v8bf*)(ar + kbB);
            af[mi].h[1] = *(const v8bf*)(ar + kbB + 8);
        }
#pragma unroll
        for (int j = 0; j < 4; ++j) {
            const bf16_t* br = bufB + (wn * 64 + j * 16 + m) * LDSPITCH_ + kbB;
            union { v16bf v; v8bf h[2]; } bfr;
            bfr.h[0] = *(const v8bf*)(br);
            bfr.h[1] = *(const v8bf*)(br + 8);
#pragma unroll
            for (int mi = 0; mi < 2; ++mi)
                acc[mi][j] = wmma_bf16(af[mi].v, bfr.v, acc[mi][j]);
        }

        wait_async0();      // next-step copies have landed
        __syncthreads();    // all waves done reading current buffer
    }

#pragma unroll
    for (int mi = 0; mi < 2; ++mi)
#pragma unroll
        for (int j = 0; j < 4; ++j) {
            const int col = nBase + wn * 64 + j * 16 + m;
#pragma unroll
            for (int r = 0; r < 8; ++r) {
                const int row = mBase + wm * 32 + mi * 16 + r + half * 8;
                C[(size_t)row * N + col] = acc[mi][j][r];
            }
        }
}

// ---------------------------------------------------------------- RMSNorm + RoPE + layout
// grid: (T, 24) -> idx 0..15 q heads, 16..19 k heads, 20..23 v heads. block = 1 wave.
__global__ __launch_bounds__(32)
void norm_rope_kernel(const float* __restrict__ qkv,
                      const float* __restrict__ cosb, const float* __restrict__ sinb,
                      const float* __restrict__ qw, const float* __restrict__ kw,
                      bf16_t* __restrict__ Q, bf16_t* __restrict__ Kb,
                      bf16_t* __restrict__ Vt) {
    const int t    = blockIdx.x;
    const int idx  = blockIdx.y;
    const int lane = threadIdx.x;

    const float* row;
    if (idx < 16)      row = qkv + (size_t)t * QKVO_ + idx * HD_;
    else if (idx < 20) row = qkv + (size_t)t * QKVO_ + NH_ * HD_ + (idx - 16) * HD_;
    else               row = qkv + (size_t)t * QKVO_ + (NH_ + NKV_) * HD_ + (idx - 20) * HD_;

    float x[4];
#pragma unroll
    for (int j = 0; j < 4; ++j) x[j] = row[lane + 32 * j];

    if (idx < 20) {
        float ss = x[0]*x[0] + x[1]*x[1] + x[2]*x[2] + x[3]*x[3];
#pragma unroll
        for (int off = 1; off < 32; off <<= 1) ss += __shfl_xor(ss, off, 32);
        const float inv = rsqrtf(ss * (1.0f / 128.0f) + 1e-6f);
        const float* w = (idx < 16) ? qw : kw;
#pragma unroll
        for (int j = 0; j < 4; ++j) x[j] = x[j] * inv * w[lane + 32 * j];

        float y[4];
#pragma unroll
        for (int j = 0; j < 4; ++j) {
            const int d = lane + 32 * j;
            const float c = cosb[(size_t)t * HD_ + d];
            const float s = sinb[(size_t)t * HD_ + d];
            const float partner = x[j ^ 2];           // d<64 pairs with d+64, else d-64
            y[j] = (d < 64) ? (x[j] * c - partner * s)
                            : (x[j] * c + partner * s);
        }
        if (idx < 16) {
            bf16_t* dst = Q + ((size_t)t * NH_ + idx) * HD_;
#pragma unroll
            for (int j = 0; j < 4; ++j) dst[lane + 32 * j] = (bf16_t)y[j];
        } else {
            bf16_t* dst = Kb + ((size_t)t * NKV_ + (idx - 16)) * HD_;
#pragma unroll
            for (int j = 0; j < 4; ++j) dst[lane + 32 * j] = (bf16_t)y[j];
        }
    } else {
        const int h4 = idx - 20;                      // V: store transposed [NKV][HD][T]
#pragma unroll
        for (int j = 0; j < 4; ++j) {
            const int d = lane + 32 * j;
            Vt[((size_t)h4 * HD_ + d) * T_ + t] = (bf16_t)x[j];
        }
    }
}

// ---------------------------------------------------------------- attention
// grid: (T/16, NH/4); block = 128 (4 waves, one head each).
// Flash-style online softmax over 32-wide s-chunks within the sliding window.
__global__ __launch_bounds__(128)
void attn_kernel(const bf16_t* __restrict__ Q, const bf16_t* __restrict__ Kb,
                 const bf16_t* __restrict__ Vt, bf16_t* __restrict__ Ob) {
    __shared__ __align__(32) bf16_t pbuf[4][16][32];   // per-wave P repack buffer

    const int lane  = threadIdx.x & 31;
    const int wave  = threadIdx.x >> 5;
    const int qtile = blockIdx.x;
    const int head  = blockIdx.y * 4 + wave;
    const int h4    = head >> 2;                       // GQA group
    const int t0    = qtile * 16;
    const int m     = lane & 15;
    const int half  = lane >> 4;
    const int kbA   = half ? 8 : 0;
    const int kbB   = half ? 16 : 0;

    // Q fragments for this 16-row tile: 4 x (16x32) covering d = 0..127
    v16bf qf[4];
    const bf16_t* qRow = Q + ((size_t)(t0 + m) * NH_ + head) * HD_;
#pragma unroll
    for (int j = 0; j < 4; ++j) {
        union { v16bf v; v8bf h[2]; } u;
        u.h[0] = *(const v8bf*)(qRow + j * 32 + kbB);
        u.h[1] = *(const v8bf*)(qRow + j * 32 + kbB + 8);
        qf[j] = u.v;
    }

    float mrow[8], lrow[8];
    v8f o[8] = {};
#pragma unroll
    for (int r = 0; r < 8; ++r) { mrow[r] = -3.0e38f; lrow[r] = 0.0f; }

    int sLo = t0 - (WINDOW_ - 1); if (sLo < 0) sLo = 0;
    const int sBase = sLo & ~31;
    const int sHi   = (((t0 + 15) >> 2) << 2) + 3;     // end of the t-tile's last 4-block

    for (int s0 = sBase; s0 <= sHi; s0 += 32) {
        // ---- S = Q @ K^T for 32 columns (two 16-wide n-tiles)
        v8f sa = {}, sb = {};
        const bf16_t* kRow0 = Kb + ((size_t)(s0 + m)      * NKV_ + h4) * HD_;
        const bf16_t* kRow1 = Kb + ((size_t)(s0 + 16 + m) * NKV_ + h4) * HD_;
#pragma unroll
        for (int j = 0; j < 4; ++j) {
            v16bf b0 = *(const v16bf*)(kRow0 + j * 32 + kbB);
            v16bf b1 = *(const v16bf*)(kRow1 + j * 32 + kbB);
            sa = wmma_bf16(qf[j], b0, sa);
            sb = wmma_bf16(qf[j], b1, sb);
        }

        // ---- mask + online softmax (rows live per lane-half in C layout)
        float p0a[8], p1a[8], scl[8];
#pragma unroll
        for (int r = 0; r < 8; ++r) {
            const int t  = t0 + r + half * 8;
            const int c0 = s0 + m;
            const int c1 = s0 + 16 + m;
            float v0 = sa[r] * SCALE_;
            float v1 = sb[r] * SCALE_;
            const bool a0 = ((c0 >> 2) <= (t >> 2)) && (t - c0 < WINDOW_);
            const bool a1 = ((c1 >> 2) <= (t >> 2)) && (t - c1 < WINDOW_);
            v0 = a0 ? v0 : -3.0e38f;
            v1 = a1 ? v1 : -3.0e38f;
            float cmax = fmaxf(v0, v1);
#pragma unroll
            for (int off = 1; off < 16; off <<= 1)
                cmax = fmaxf(cmax, __shfl_xor(cmax, off, 32));
            const float mnew = fmaxf(mrow[r], cmax);
            float p0 = a0 ? __expf(v0 - mnew) : 0.0f;
            float p1 = a1 ? __expf(v1 - mnew) : 0.0f;
            float rs = p0 + p1;
#pragma unroll
            for (int off = 1; off < 16; off <<= 1) rs += __shfl_xor(rs, off, 32);
            const float sc = __expf(mrow[r] - mnew);
            lrow[r] = lrow[r] * sc + rs;
            mrow[r] = mnew;
            scl[r]  = sc;
            p0a[r]  = p0;
            p1a[r]  = p1;
        }
#pragma unroll
        for (int dt = 0; dt < 8; ++dt)
#pragma unroll
            for (int r = 0; r < 8; ++r) o[dt][r] *= scl[r];

        // ---- repack P (C layout) -> A layout via per-wave LDS bounce
#pragma unroll
        for (int r = 0; r < 8; ++r) {
            const int row = r + half * 8;
            pbuf[wave][row][m]      = (bf16_t)p0a[r];
            pbuf[wave][row][m + 16] = (bf16_t)p1a[r];
        }
        asm volatile("s_wait_dscnt 0" ::: "memory");   // same-wave LDS ordering
        union { v16bf v; v8bf h[2]; } up;
        up.h[0] = *(const v8bf*)&pbuf[wave][m][kbB];
        up.h[1] = *(const v8bf*)&pbuf[wave][m][kbB + 8];

        // ---- O += P @ V  (V transposed: s contiguous per d-row)
#pragma unroll
        for (int dt = 0; dt < 8; ++dt) {
            const bf16_t* vRow = Vt + ((size_t)h4 * HD_ + dt * 16 + m) * T_ + s0 + kbB;
            v16bf bv = *(const v16bf*)vRow;
            o[dt] = wmma_bf16(up.v, bv, o[dt]);
        }
    }

    // ---- epilogue: normalize and store bf16 [T][NH*HD]
#pragma unroll
    for (int r = 0; r < 8; ++r) {
        const int row = r + half * 8;
        const int t   = t0 + row;
        const float inv = 1.0f / lrow[r];
#pragma unroll
        for (int dt = 0; dt < 8; ++dt)
            Ob[((size_t)t * NH_ + head) * HD_ + dt * 16 + m] = (bf16_t)(o[dt][r] * inv);
    }
}

// ---------------------------------------------------------------- launch
extern "C" void kernel_launch(void* const* d_in, const int* in_sizes, int n_in,
                              void* d_out, int out_size, void* d_ws, size_t ws_size,
                              hipStream_t stream) {
    const float* hidden   = (const float*)d_in[0];
    const float* cosb     = (const float*)d_in[1];
    const float* sinb     = (const float*)d_in[2];
    const float* w_qkv    = (const float*)d_in[3];
    const float* q_norm_w = (const float*)d_in[4];
    const float* k_norm_w = (const float*)d_in[5];
    const float* w_o      = (const float*)d_in[6];
    float* out = (float*)d_out;

    char* ws = (char*)d_ws;
    size_t off = 0;
    auto carve = [&](size_t bytes) {
        char* p = ws + off;
        off = (off + bytes + 255) & ~(size_t)255;
        return p;
    };
    bf16_t* hidden_bf = (bf16_t*)carve((size_t)T_ * HIDDEN_ * 2);
    bf16_t* wqkv_bf   = (bf16_t*)carve((size_t)QKVO_ * HIDDEN_ * 2);
    bf16_t* wo_bf     = (bf16_t*)carve((size_t)HIDDEN_ * NH_ * HD_ * 2);
    float*  qkv_f32   = (float*) carve((size_t)T_ * QKVO_ * 4);
    bf16_t* Qb        = (bf16_t*)carve((size_t)T_ * NH_ * HD_ * 2);
    bf16_t* Kb        = (bf16_t*)carve((size_t)T_ * NKV_ * HD_ * 2);
    bf16_t* Vt        = (bf16_t*)carve((size_t)NKV_ * HD_ * T_ * 2);
    bf16_t* attn_bf   = (bf16_t*)carve((size_t)T_ * NH_ * HD_ * 2);

    // 1) fp32 -> bf16 conversions
    f32_to_bf16_kernel<<<4096, 256, 0, stream>>>(hidden, hidden_bf, T_ * HIDDEN_);
    f32_to_bf16_kernel<<<4096, 256, 0, stream>>>(w_qkv, wqkv_bf, QKVO_ * HIDDEN_);
    f32_to_bf16_kernel<<<4096, 256, 0, stream>>>(w_o, wo_bf, HIDDEN_ * NH_ * HD_);

    // 2) QKV projection: [T,2048] @ [3072,2048]^T -> [T,3072] fp32
    gemm_bf16_kernel<<<dim3(QKVO_ / 128, T_ / 128), 256, 0, stream>>>(
        hidden_bf, wqkv_bf, qkv_f32, T_, QKVO_, HIDDEN_);

    // 3) RMSNorm + RoPE + bf16 layout (V transposed)
    norm_rope_kernel<<<dim3(T_, NH_ + 2 * NKV_), 32, 0, stream>>>(
        qkv_f32, cosb, sinb, q_norm_w, k_norm_w, Qb, Kb, Vt);

    // 4) windowed block-causal attention
    attn_kernel<<<dim3(T_ / 16, NH_ / 4), 128, 0, stream>>>(Qb, Kb, Vt, attn_bf);

    // 5) output projection: [T,2048] @ [2048,2048]^T -> d_out fp32
    gemm_bf16_kernel<<<dim3(HIDDEN_ / 128, T_ / 128), 256, 0, stream>>>(
        attn_bf, wo_bf, out, T_, HIDDEN_, NH_ * HD_);
}
